// GraphAttention_44813688766608
// MI455X (gfx1250) — matrix operand order repaired and checked by reading, not compiled, not run on hardware
//
#include <hip/hip_runtime.h>
#include <hip/hip_bf16.h>

// Sizes (fixed by the reference)
#define Bsz   2
#define Nsz   2048
#define Fin   128
#define Usz   64
#define Hsz   8

typedef __attribute__((ext_vector_type(16))) _Float16 v16h;
typedef __attribute__((ext_vector_type(8)))  _Float16 v8h;
typedef __attribute__((ext_vector_type(8)))  float    v8f;

#define NEGBIG (-3.402823466e38f)

__device__ __forceinline__ float lrelu(float v) { return v > 0.0f ? v : 0.2f * v; }

// ---------------------------------------------------------------------------
// Prep 1: w_src[h][f] = sum_u kernels[h][f][u] * att[h][u]
//         w_dst[h][f] = sum_u kernels[h][f][u] * att[h][U+u]
// ---------------------------------------------------------------------------
__global__ void prep_w_kernel(const float* __restrict__ kern,
                              const float* __restrict__ attk,
                              float* __restrict__ wsrc,
                              float* __restrict__ wdst) {
    int t = blockIdx.x * blockDim.x + threadIdx.x;
    if (t >= Hsz * Fin) return;
    int h = t >> 7, f = t & 127;
    const float* kp = kern + ((size_t)h * Fin + f) * Usz;
    const float* ap = attk + (size_t)h * (2 * Usz);
    float s = 0.f, d = 0.f;
#pragma unroll 8
    for (int u = 0; u < Usz; ++u) {
        float kv = kp[u];
        s += kv * ap[u];
        d += kv * ap[Usz + u];
    }
    wsrc[t] = s;
    wdst[t] = d;
}

// ---------------------------------------------------------------------------
// Prep 2: src[h][b][n] = x[b][n][:] . w_src[h][:]   (and dst likewise)
// ---------------------------------------------------------------------------
__global__ void prep_srcdst_kernel(const float* __restrict__ x,
                                   const float* __restrict__ wsrc,
                                   const float* __restrict__ wdst,
                                   float* __restrict__ srcg,
                                   float* __restrict__ dstg) {
    int t = blockIdx.x * blockDim.x + threadIdx.x;   // 0 .. H*B*N-1
    if (t >= Hsz * Bsz * Nsz) return;
    int n = t & (Nsz - 1);
    int b = (t >> 11) & (Bsz - 1);
    int h = t >> 12;
    const float* xr = x + ((size_t)b * Nsz + n) * Fin;
    const float* ws = wsrc + (size_t)h * Fin;
    const float* wd = wdst + (size_t)h * Fin;
    float s = 0.f, d = 0.f;
#pragma unroll 8
    for (int f = 0; f < Fin; ++f) {
        float xv = xr[f];
        s += xv * ws[f];
        d += xv * wd[f];
    }
    srcg[t] = s;
    dstg[t] = d;
}

// ---------------------------------------------------------------------------
// Prep 3: xhT[b][f][j] = (f16) x[b][j][f]   (transpose so B-fragments are
//         16 contiguous f16 -> global_load_b128 pairs)
// ---------------------------------------------------------------------------
__global__ void prep_xt_kernel(const float* __restrict__ x,
                               _Float16* __restrict__ xht) {
    int t = blockIdx.x * blockDim.x + threadIdx.x;   // 0 .. B*F*N-1
    if (t >= Bsz * Fin * Nsz) return;
    int j = t & (Nsz - 1);
    int f = (t >> 11) & (Fin - 1);
    int b = t >> 18;
    xht[t] = (_Float16)x[((size_t)b * Nsz + j) * Fin + f];
}

// ---------------------------------------------------------------------------
// Main: per (h, b, 16-row tile): masked softmax over j, then attn @ x via
//       v_wmma_f32_16x16x32_f16.  Block = 8 waves = 8 row tiles (128 rows).
// ---------------------------------------------------------------------------
__global__ void __launch_bounds__(256)
gat_main_kernel(const int* __restrict__ a,
                const float* __restrict__ srcg,
                const float* __restrict__ dstg,
                const _Float16* __restrict__ xht,
                float* __restrict__ out) {
    __shared__ float sdst[Nsz];            // dst[h][b][:]   (8 KB)
    __shared__ float ssrc[128];            // src rows of this slab
    __shared__ float smax[8][16];
    __shared__ float sinv[8][16];
    __shared__ __align__(32) _Float16 satile[8][16 * 32];  // per-wave A stage

    const int ib = blockIdx.x & 15;                 // row-slab (128 rows)
    const int b  = (blockIdx.x >> 4) & 1;
    const int h  = blockIdx.x >> 5;
    const int wv   = threadIdx.x >> 5;
    const int lane = threadIdx.x & 31;
    const int i0 = ib * 128 + wv * 16;              // first row of this wave

    // stage dst[h][b][:] and src rows into LDS
    const int hb = (h * Bsz + b) * Nsz;
    for (int t = threadIdx.x; t < Nsz; t += 256) sdst[t] = dstg[hb + t];
    for (int t = threadIdx.x; t < 128; t += 256) ssrc[t] = srcg[hb + ib * 128 + t];
    __syncthreads();

    const int* ag = a + (size_t)b * Nsz * Nsz;

    // ---- Pass 1: per-row masked softmax stats (two-pass: max, then sum) ----
    for (int m = 0; m < 16; ++m) {
        const float srcm = ssrc[(wv << 4) + m];
        const int2* ap = (const int2*)(ag + ((size_t)(i0 + m) << 11));
        float mx = NEGBIG;
        for (int t = lane; t < Nsz / 2; t += 32) {
            int2 av = ap[t];
            float e0 = av.x ? lrelu(srcm + sdst[2 * t])     : NEGBIG;
            float e1 = av.y ? lrelu(srcm + sdst[2 * t + 1]) : NEGBIG;
            mx = fmaxf(mx, fmaxf(e0, e1));
        }
#pragma unroll
        for (int off = 16; off; off >>= 1) mx = fmaxf(mx, __shfl_xor(mx, off, 32));
        float sum = 0.f;
        for (int t = lane; t < Nsz / 2; t += 32) {
            int2 av = ap[t];
            float e0 = av.x ? lrelu(srcm + sdst[2 * t])     : NEGBIG;
            float e1 = av.y ? lrelu(srcm + sdst[2 * t + 1]) : NEGBIG;
            sum += __expf(e0 - mx) + __expf(e1 - mx);
        }
#pragma unroll
        for (int off = 16; off; off >>= 1) sum += __shfl_xor(sum, off, 32);
        if (lane == m) { smax[wv][m] = mx; sinv[wv][m] = 1.0f / sum; }
    }
    asm volatile("s_wait_dscnt 0" ::: "memory");

    // ---- Pass 2: stream 32-wide k-steps, 8 feature tiles of WMMA each ----
    v8f acc[8];
#pragma unroll
    for (int i = 0; i < 8; ++i) acc[i] = (v8f)0.0f;

    const _Float16* xb = xht + (size_t)b * Fin * Nsz;
    const int c0    = (lane & 15) << 1;          // p-compute: 2 cols
    const int rbase = (lane >> 4) << 3;          // p-compute: 8 rows
    const int mA    = lane & 15;                 // A-frag row
    const int kbA   = (lane < 16) ? 0 : 16;      // A-frag K base
    const int nf    = lane & 15;                 // B-frag column
    const int kbB   = (lane < 16) ? 0 : 16;      // B-frag K base

    for (int ks = 0; ks < Nsz / 32; ++ks) {
        const int jbase = ks << 5;
        // compute p tile (16 x 32) into per-wave LDS, row-major f16
        const float d0 = sdst[jbase + c0];
        const float d1 = sdst[jbase + c0 + 1];
#pragma unroll
        for (int rr = 0; rr < 8; ++rr) {
            const int m = rbase + rr;
            const int2 av = *(const int2*)(ag + ((size_t)(i0 + m) << 11) + jbase + c0);
            const float srcm = ssrc[(wv << 4) + m];
            const float rm = smax[wv][m];
            const float ri = sinv[wv][m];
            float e0 = av.x ? lrelu(srcm + d0) : NEGBIG;
            float e1 = av.y ? lrelu(srcm + d1) : NEGBIG;
            float p0 = __expf(e0 - rm) * ri;
            float p1 = __expf(e1 - rm) * ri;
            union { _Float16 hh[2]; unsigned u; } pk;
            pk.hh[0] = (_Float16)p0;
            pk.hh[1] = (_Float16)p1;
            *(unsigned*)&satile[wv][(m << 5) + c0] = pk.u;
        }
        asm volatile("s_wait_dscnt 0" ::: "memory");

        // A fragment (contiguous K, matching B): lane<16 -> K 0-15; lane>=16 -> K 16-31
        v8h alo = *(const v8h*)&satile[wv][(mA << 5) + kbA];
        v8h ahi = *(const v8h*)&satile[wv][(mA << 5) + kbA + 8];
        v16h af;
#pragma unroll
        for (int i = 0; i < 8; ++i) { af[i] = alo[i]; af[i + 8] = ahi[i]; }

        // B fragments (16 contiguous f16 from xhT) + WMMA, 8 feature tiles
#pragma unroll
        for (int ft = 0; ft < 8; ++ft) {
            v16h bf = *(const v16h*)(xb + (((size_t)(ft << 4) + nf) << 11) + jbase + kbB);
            acc[ft] = __builtin_amdgcn_wmma_f32_16x16x32_f16(
                false, af, false, bf, (short)0, acc[ft], false, false);
        }
        asm volatile("s_wait_dscnt 0" ::: "memory");
    }

    // ---- Epilogue: ReLU + store. C layout: VGPR r -> row (r + 8*(lane>=16)),
    //      col = lane & 15.  out[b][i][h*128 + f], row stride H*F = 1024. ----
    float* op = out + (((size_t)b * Nsz + i0) << 10) + (h << 7);
    const int mb = (lane < 16) ? 0 : 8;
#pragma unroll
    for (int ft = 0; ft < 8; ++ft) {
#pragma unroll
        for (int r = 0; r < 8; ++r) {
            float v = acc[ft][r];
            op[((size_t)(mb + r) << 10) + (ft << 4) + nf] = v > 0.0f ? v : 0.0f;
        }
    }
}

// ---------------------------------------------------------------------------
// Launch
// ---------------------------------------------------------------------------
extern "C" void kernel_launch(void* const* d_in, const int* in_sizes, int n_in,
                              void* d_out, int out_size, void* d_ws, size_t ws_size,
                              hipStream_t stream) {
    const float* x    = (const float*)d_in[0];   // [B,N,F]
    const int*   a    = (const int*)d_in[1];     // [B,N,N]
    const float* kern = (const float*)d_in[2];   // [H,F,U]
    const float* attk = (const float*)d_in[3];   // [H,2U]
    float* out = (float*)d_out;                  // [B,N,H*F]

    // workspace layout
    char* ws = (char*)d_ws;
    float*    wsrc = (float*)(ws + 0);                     //  4 KB
    float*    wdst = (float*)(ws + 4096);                  //  4 KB
    float*    srcg = (float*)(ws + 8192);                  // 128 KB
    float*    dstg = (float*)(ws + 8192 + 131072);         // 128 KB
    _Float16* xht  = (_Float16*)(ws + 8192 + 2 * 131072);  //   1 MB

    prep_w_kernel<<<(Hsz * Fin + 255) / 256, 256, 0, stream>>>(kern, attk, wsrc, wdst);
    prep_srcdst_kernel<<<(Hsz * Bsz * Nsz + 255) / 256, 256, 0, stream>>>(x, wsrc, wdst, srcg, dstg);
    prep_xt_kernel<<<(Bsz * Fin * Nsz + 255) / 256, 256, 0, stream>>>(x, xht);
    gat_main_kernel<<<Hsz * Bsz * (Nsz / 128), 256, 0, stream>>>(a, srcg, dstg, xht, out);
}
